// MultiHeadAttention_25168508355023
// MI455X (gfx1250) — matrix operand order repaired and checked by reading, not compiled, not run on hardware
//
#include <hip/hip_runtime.h>
#include <hip/hip_bf16.h>

typedef __attribute__((ext_vector_type(16))) __bf16 v16bf;
typedef __attribute__((ext_vector_type(8)))  __bf16 v8bf;
typedef __attribute__((ext_vector_type(8)))  float  v8f;

#define H_  16
#define DK_ 64
#define DV_ 64
#define D_  1024
#define B_  4
#define S_  2048
#define BS_ (B_*S_)

#define WMMA_BF16(a, b, c) \
  __builtin_amdgcn_wmma_f32_16x16x32_bf16(false, (a), false, (b), (short)0, (c), false, false)

// sched_group_barrier masks
#define SG_WMMA   0x008
#define SG_VMREAD 0x020

__device__ __forceinline__ __bf16 f2bf(float f) {
  unsigned u = __builtin_bit_cast(unsigned, f);
  unsigned r = u + 0x7FFFu + ((u >> 16) & 1u);   // round-to-nearest-even
  unsigned short hx = (unsigned short)(r >> 16);
  return __builtin_bit_cast(__bf16, hx);
}

__device__ __forceinline__ v16bf mk16(v8bf lo, v8bf hi) {
  v16bf r;
#pragma unroll
  for (int i = 0; i < 8; ++i) { r[i] = lo[i]; r[i + 8] = hi[i]; }
  return r;
}

// Pin a software-pipelined shape on a GEMM step: interleave VMEM reads with WMMAs
__device__ __forceinline__ void gemm_pipeline_hint() {
#pragma unroll
  for (int i = 0; i < 4; ++i) {
    __builtin_amdgcn_sched_group_barrier(SG_VMREAD, 5, 0);  // 5 fragment loads
    __builtin_amdgcn_sched_group_barrier(SG_WMMA,   4, 0);  // 4 WMMAs
  }
}

// ---------------- prep: fp32 -> bf16 convert ----------------
__global__ void k_f32_to_bf16(const float* __restrict__ src, __bf16* __restrict__ dst, int n) {
  int i = blockIdx.x * blockDim.x + threadIdx.x;
  int stride = gridDim.x * blockDim.x;
  for (; i < n; i += stride) dst[i] = f2bf(src[i]);
}

// ---------------- prep: pack per-head weight [H][D][64] into B-fragment order ----------------
// packed index = (((h*32 + kt)*4 + nt)*32 + lane)*16 + e
// lane L: n = nt*16 + (L&15); k = kt*32 + (L>>4)*16 + e  (contiguous 16-K run per lane)
__global__ void k_pack_w_head(const float* __restrict__ w, __bf16* __restrict__ out) {
  int i = blockIdx.x * blockDim.x + threadIdx.x;
  int stride = gridDim.x * blockDim.x;
  const int total = H_ * D_ * DK_;           // 1M elements
  for (; i < total; i += stride) {
    int e    = i & 15;
    int lane = (i >> 4) & 31;
    int nt   = (i >> 9) & 3;
    int kt   = (i >> 11) & 31;
    int h    = (i >> 16);
    int n = nt * 16 + (lane & 15);
    int k = kt * 32 + (lane >> 4) * 16 + e;
    out[i] = f2bf(w[((size_t)h * D_ + k) * DK_ + n]);
  }
}

// ---------------- prep: pack Wo [1024][1024] into B-fragment order ----------------
// packed index = ((kt*64 + nt)*32 + lane)*16 + e
__global__ void k_pack_wo(const float* __restrict__ w, __bf16* __restrict__ out) {
  int i = blockIdx.x * blockDim.x + threadIdx.x;
  int stride = gridDim.x * blockDim.x;
  const int total = D_ * D_;                 // 1M elements
  for (; i < total; i += stride) {
    int e    = i & 15;
    int lane = (i >> 4) & 31;
    int nt   = (i >> 9) & 63;
    int kt   = (i >> 15);
    int n = nt * 16 + (lane & 15);
    int k = kt * 32 + (lane >> 4) * 16 + e;
    out[i] = f2bf(w[(size_t)k * D_ + n]);
  }
}

// ---------------- QKV projection: one wave computes a 16x64 strip for one (m-tile, head, sel) ----------------
__global__ __launch_bounds__(32)
void k_qkv(const __bf16* __restrict__ xb,
           const __bf16* __restrict__ wqp, const __bf16* __restrict__ wkp, const __bf16* __restrict__ wvp,
           const float* __restrict__ bq, const float* __restrict__ bk, const float* __restrict__ bv,
           __bf16* __restrict__ Qb, __bf16* __restrict__ Kb, __bf16* __restrict__ Vt) {
  const int lane = threadIdx.x;
  const int nl = lane & 15, hh = lane >> 4;
  const int mt  = blockIdx.x;       // 0..511 (rows of flattened B*S)
  const int h   = blockIdx.y;       // head
  const int sel = blockIdx.z;       // 0:Q 1:K 2:V

  const __bf16* wp   = (sel == 0) ? wqp : ((sel == 1) ? wkp : wvp);
  const float*  bias = (sel == 0) ? bq  : ((sel == 1) ? bk  : bv);

  const int m0 = mt * 16;
  const __bf16* arow = xb + (size_t)(m0 + nl) * D_;
  const __bf16* wph  = wp + (size_t)h * (D_ * DK_);

  auto ldA = [&](int kt) -> v16bf {
    return mk16(*(const v8bf*)(arow + kt * 32 + hh * 16),
                *(const v8bf*)(arow + kt * 32 + hh * 16 + 8));
  };
  auto ldB = [&](int kt, int nt) -> v16bf {
    return *(const v16bf*)(wph + (size_t)kt * (4 * 32 * 16) + ((size_t)nt * 32 + lane) * 16);
  };

  v8f acc[4] = { {}, {}, {}, {} };

  // software-pipelined K loop: buffer set A computes while set B loads (and vice versa)
  v16bf aA  = ldA(0);
  v16bf bA0 = ldB(0, 0), bA1 = ldB(0, 1), bA2 = ldB(0, 2), bA3 = ldB(0, 3);
  for (int kt = 0; kt < D_ / 32; kt += 2) {
    v16bf aB  = ldA(kt + 1);
    v16bf bB0 = ldB(kt + 1, 0), bB1 = ldB(kt + 1, 1), bB2 = ldB(kt + 1, 2), bB3 = ldB(kt + 1, 3);
    acc[0] = WMMA_BF16(aA, bA0, acc[0]);
    acc[1] = WMMA_BF16(aA, bA1, acc[1]);
    acc[2] = WMMA_BF16(aA, bA2, acc[2]);
    acc[3] = WMMA_BF16(aA, bA3, acc[3]);
    int kn = (kt + 2) & (D_ / 32 - 1);        // wraps to 0 on the last trip (dead load)
    aA  = ldA(kn);
    bA0 = ldB(kn, 0); bA1 = ldB(kn, 1); bA2 = ldB(kn, 2); bA3 = ldB(kn, 3);
    acc[0] = WMMA_BF16(aB, bB0, acc[0]);
    acc[1] = WMMA_BF16(aB, bB1, acc[1]);
    acc[2] = WMMA_BF16(aB, bB2, acc[2]);
    acc[3] = WMMA_BF16(aB, bB3, acc[3]);
    gemm_pipeline_hint();   // force loads to interleave ahead of consuming WMMAs
  }

#pragma unroll
  for (int nt = 0; nt < 4; ++nt) {
    float bval = bias[h * DK_ + nt * 16 + nl];
#pragma unroll
    for (int r = 0; r < 8; ++r) acc[nt][r] += bval;
  }

  const int bidx  = m0 / S_;
  const int sbase = m0 % S_;
  const size_t bh = (size_t)bidx * H_ + h;

  if (sel == 2) {
    // V stored transposed: Vt[b][h][v][s]; lane's 8 rows are contiguous in s -> one 16B store
#pragma unroll
    for (int nt = 0; nt < 4; ++nt) {
      v8bf pk;
#pragma unroll
      for (int r = 0; r < 8; ++r) pk[r] = f2bf(acc[nt][r]);
      int v = nt * 16 + nl;
      *(v8bf*)(Vt + (bh * DV_ + v) * S_ + sbase + hh * 8) = pk;
    }
  } else {
    __bf16* outp = (sel == 0) ? Qb : Kb;
#pragma unroll
    for (int nt = 0; nt < 4; ++nt) {
      int nn = nt * 16 + nl;
#pragma unroll
      for (int r = 0; r < 8; ++r) {
        int s = sbase + 8 * hh + r;       // D-layout: M = r + 8*(lane>>4)
        outp[(bh * S_ + s) * DK_ + nn] = f2bf(acc[nt][r]);
      }
    }
  }
}

// ---------------- flash attention: one wave per (16-query block, head, batch) ----------------
__global__ __launch_bounds__(32)
void k_attn(const __bf16* __restrict__ Qb, const __bf16* __restrict__ Kb,
            const __bf16* __restrict__ Vt, __bf16* __restrict__ cat) {
  __shared__ __bf16 lds_p[16 * 32];       // P tile staged D-layout -> A-layout
  const int lane = threadIdx.x;
  const int nl = lane & 15, hh = lane >> 4;
  const int st = blockIdx.x;              // query tile 0..127
  const int h  = blockIdx.y;
  const int b  = blockIdx.z;
  const size_t bh = (size_t)b * H_ + h;
  const int s0 = st * 16;

  // Q A-fragments (dk = 0..31 and 32..63), held for the whole loop
  const __bf16* qrow = Qb + (bh * S_ + s0 + nl) * DK_;
  v16bf qa0 = mk16(*(const v8bf*)(qrow + hh * 16),      *(const v8bf*)(qrow + 8 + hh * 16));
  v16bf qa1 = mk16(*(const v8bf*)(qrow + 32 + hh * 16), *(const v8bf*)(qrow + 40 + hh * 16));

  const __bf16* vhead = Vt + bh * DV_ * S_;

  auto ldK = [&](int t, int kthalf) -> v16bf {
    return *(const v16bf*)(Kb + (bh * S_ + t + nl) * DK_ + kthalf * 32 + hh * 16);
  };
  auto ldV = [&](int t0, int nt) -> v16bf {
    return *(const v16bf*)(vhead + (size_t)(nt * 16 + nl) * S_ + t0 + hh * 16);
  };

  v8f o0 = {}, o1 = {}, o2 = {}, o3 = {};
  float mrow[8], lrow[8];
#pragma unroll
  for (int r = 0; r < 8; ++r) { mrow[r] = -3.0e38f; lrow[r] = 0.0f; }

  const float scale = 0.125f;             // 1/sqrt(64)

  // preload first K tile's B-fragments
  v16bf kb0 = ldK(0, 0), kb1 = ldK(0, 1), kb2 = ldK(16, 0), kb3 = ldK(16, 1);

#pragma unroll 2
  for (int t0 = 0; t0 < S_; t0 += 32) {
    // ---- scores: two 16x16 tiles over 32 keys, K-dim = dk = 64 (2 WMMAs each) ----
    v8f c0 = {}, c1 = {};
    c0 = WMMA_BF16(qa0, kb0, c0);
    c0 = WMMA_BF16(qa1, kb1, c0);
    c1 = WMMA_BF16(qa0, kb2, c1);
    c1 = WMMA_BF16(qa1, kb3, c1);

    // ---- issue next-tile K loads and current-tile V loads NOW; softmax VALU hides them ----
    int tn = (t0 + 32) & (S_ - 1);        // wraps to 0 on the last trip (dead load)
    v16bf nk0 = ldK(tn, 0), nk1 = ldK(tn, 1), nk2 = ldK(tn + 16, 0), nk3 = ldK(tn + 16, 1);
    v16bf vb0 = ldV(t0, 0), vb1 = ldV(t0, 1), vb2 = ldV(t0, 2), vb3 = ldV(t0, 3);
    // keep the loads above this point; keep the softmax below it
    __builtin_amdgcn_sched_barrier(0);

    // ---- online softmax over the 32 columns (rows live per lane-half) ----
#pragma unroll
    for (int r = 0; r < 8; ++r) {
      c0[r] *= scale; c1[r] *= scale;
      float tv = fmaxf(c0[r], c1[r]);
#pragma unroll
      for (int msk = 1; msk < 16; msk <<= 1) tv = fmaxf(tv, __shfl_xor(tv, msk, 32));
      float mnew  = fmaxf(mrow[r], tv);
      float alpha = __expf(mrow[r] - mnew);
      mrow[r] = mnew;
      float p0 = __expf(c0[r] - mnew);
      float p1 = __expf(c1[r] - mnew);
      float ts = p0 + p1;
#pragma unroll
      for (int msk = 1; msk < 16; msk <<= 1) ts += __shfl_xor(ts, msk, 32);
      lrow[r] = lrow[r] * alpha + ts;
      o0[r] *= alpha; o1[r] *= alpha; o2[r] *= alpha; o3[r] *= alpha;
      // D-layout (row r+8*hh, col nl) -> row-major LDS tile
      lds_p[(r + 8 * hh) * 32 + nl]      = f2bf(p0);
      lds_p[(r + 8 * hh) * 32 + 16 + nl] = f2bf(p1);
    }

    // ---- P A-fragment from LDS (DS ops are in-order within this single wave) ----
    const __bf16* prow = &lds_p[nl * 32];
    v16bf pa = mk16(*(const v8bf*)(prow + hh * 16), *(const v8bf*)(prow + hh * 16 + 8));

    // ---- PV: 4 dv-tiles, B-frags contiguous in t thanks to transposed V ----
    o0 = WMMA_BF16(pa, vb0, o0);
    o1 = WMMA_BF16(pa, vb1, o1);
    o2 = WMMA_BF16(pa, vb2, o2);
    o3 = WMMA_BF16(pa, vb3, o3);

    // rotate double-buffered K fragments
    kb0 = nk0; kb1 = nk1; kb2 = nk2; kb3 = nk3;
  }

  // ---- normalize and write concat-layout bf16 for the output projection ----
#pragma unroll
  for (int r = 0; r < 8; ++r) {
    float inv = 1.0f / lrow[r];
    int s = s0 + 8 * hh + r;
    __bf16* crow = cat + ((size_t)b * S_ + s) * D_ + h * DV_;
    crow[0 * 16 + nl] = f2bf(o0[r] * inv);
    crow[1 * 16 + nl] = f2bf(o1[r] * inv);
    crow[2 * 16 + nl] = f2bf(o2[r] * inv);
    crow[3 * 16 + nl] = f2bf(o3[r] * inv);
  }
}

// ---------------- output projection: one wave computes a 16x64 strip of [8192 x 1024] ----------------
__global__ __launch_bounds__(32)
void k_oproj(const __bf16* __restrict__ cat, const __bf16* __restrict__ wop,
             const float* __restrict__ bo, float* __restrict__ outp) {
  const int lane = threadIdx.x;
  const int nl = lane & 15, hh = lane >> 4;
  const int mt = blockIdx.x;   // 0..511
  const int ng = blockIdx.y;   // 0..15 (strip of 4 n-tiles)
  const int m0 = mt * 16;
  const __bf16* arow = cat + (size_t)(m0 + nl) * D_;

  auto ldA = [&](int kt) -> v16bf {
    return mk16(*(const v8bf*)(arow + kt * 32 + hh * 16),
                *(const v8bf*)(arow + kt * 32 + hh * 16 + 8));
  };
  auto ldB = [&](int kt, int nt) -> v16bf {
    return *(const v16bf*)(wop + ((size_t)kt * 64 + (size_t)ng * 4 + nt) * (32 * 16)
                               + (size_t)lane * 16);
  };

  v8f acc[4] = { {}, {}, {}, {} };

  v16bf aA  = ldA(0);
  v16bf bA0 = ldB(0, 0), bA1 = ldB(0, 1), bA2 = ldB(0, 2), bA3 = ldB(0, 3);
  for (int kt = 0; kt < D_ / 32; kt += 2) {
    v16bf aB  = ldA(kt + 1);
    v16bf bB0 = ldB(kt + 1, 0), bB1 = ldB(kt + 1, 1), bB2 = ldB(kt + 1, 2), bB3 = ldB(kt + 1, 3);
    acc[0] = WMMA_BF16(aA, bA0, acc[0]);
    acc[1] = WMMA_BF16(aA, bA1, acc[1]);
    acc[2] = WMMA_BF16(aA, bA2, acc[2]);
    acc[3] = WMMA_BF16(aA, bA3, acc[3]);
    int kn = (kt + 2) & (D_ / 32 - 1);        // wraps to 0 on the last trip (dead load)
    aA  = ldA(kn);
    bA0 = ldB(kn, 0); bA1 = ldB(kn, 1); bA2 = ldB(kn, 2); bA3 = ldB(kn, 3);
    acc[0] = WMMA_BF16(aB, bB0, acc[0]);
    acc[1] = WMMA_BF16(aB, bB1, acc[1]);
    acc[2] = WMMA_BF16(aB, bB2, acc[2]);
    acc[3] = WMMA_BF16(aB, bB3, acc[3]);
    gemm_pipeline_hint();   // force loads to interleave ahead of consuming WMMAs
  }

#pragma unroll
  for (int nt = 0; nt < 4; ++nt) {
    int n = (ng * 4 + nt) * 16 + nl;
    float bval = bo[n];
#pragma unroll
    for (int r = 0; r < 8; ++r) {
      outp[(size_t)(m0 + 8 * hh + r) * D_ + n] = acc[nt][r] + bval;
    }
  }
}

extern "C" void kernel_launch(void* const* d_in, const int* in_sizes, int n_in,
                              void* d_out, int out_size, void* d_ws, size_t ws_size,
                              hipStream_t stream) {
  (void)in_sizes; (void)n_in; (void)out_size; (void)ws_size;
  const float* x  = (const float*)d_in[0];
  const float* Wq = (const float*)d_in[1];
  const float* bq = (const float*)d_in[2];
  const float* Wk = (const float*)d_in[3];
  const float* bk = (const float*)d_in[4];
  const float* Wv = (const float*)d_in[5];
  const float* bv = (const float*)d_in[6];
  const float* Wo = (const float*)d_in[7];
  const float* bo = (const float*)d_in[8];
  float* outp = (float*)d_out;

  char* ws = (char*)d_ws;
  const size_t MB = (size_t)1 << 20;
  __bf16* xb  = (__bf16*)(ws);            // 16 MB : x in bf16        [BS][D]
  __bf16* wqp = (__bf16*)(ws + 16 * MB);  //  2 MB : packed Wq frags
  __bf16* wkp = (__bf16*)(ws + 18 * MB);  //  2 MB : packed Wk frags
  __bf16* wvp = (__bf16*)(ws + 20 * MB);  //  2 MB : packed Wv frags
  __bf16* wop = (__bf16*)(ws + 22 * MB);  //  2 MB : packed Wo frags
  __bf16* Qb  = (__bf16*)(ws + 24 * MB);  // 16 MB : Q  [B][H][S][dk]
  __bf16* Kb  = (__bf16*)(ws + 40 * MB);  // 16 MB : K  [B][H][S][dk]
  __bf16* Vt  = (__bf16*)(ws + 56 * MB);  // 16 MB : V  [B][H][dv][S] (transposed)
  __bf16* cat = (__bf16*)(ws + 72 * MB);  // 16 MB : concat heads [B][S][H*dv]

  k_f32_to_bf16<<<4096, 256, 0, stream>>>(x, xb, BS_ * D_);
  k_pack_w_head<<<1024, 256, 0, stream>>>(Wq, wqp);
  k_pack_w_head<<<1024, 256, 0, stream>>>(Wk, wkp);
  k_pack_w_head<<<1024, 256, 0, stream>>>(Wv, wvp);
  k_pack_wo<<<1024, 256, 0, stream>>>(Wo, wop);

  k_qkv<<<dim3(BS_ / 16, H_, 3), 32, 0, stream>>>(xb, wqp, wkp, wvp, bq, bk, bv, Qb, Kb, Vt);
  k_attn<<<dim3(S_ / 16, H_, B_), 32, 0, stream>>>(Qb, Kb, Vt, cat);
  k_oproj<<<dim3(BS_ / 16, 16), 32, 0, stream>>>(cat, wop, bo, outp);
}
